// MultiSAGENet_8143257993951
// MI455X (gfx1250) — compile-verified, matrix operand order checked by reading, not run on hardware
//
#include <hip/hip_runtime.h>
#include <hip/hip_bf16.h>
#include <math.h>

typedef __attribute__((ext_vector_type(16))) _Float16 v16h;
typedef __attribute__((ext_vector_type(8)))  _Float16 v8h;
typedef __attribute__((ext_vector_type(4)))  _Float16 v4h;
typedef __attribute__((ext_vector_type(8)))  float    v8f;

#define POOLW 68     // per-graph accumulator width: 4 (input) + 8 layers * 8
#define ROWW  204    // pooled row width (logical): 3 * POOLW
#define ROWP  224    // pooled row padded to 7 WMMA K-chunks of 32
#define GWIN  64     // LDS pooling window (graphs per block window)
#define USE_ASYNC_LDS 1

// ---------- monotonic float<->uint key for atomic max (graph pooling only) ----------
__device__ __forceinline__ unsigned fmax_key(float f) {
    unsigned u = __float_as_uint(f);
    return (u & 0x80000000u) ? ~u : (u | 0x80000000u);
}
__device__ __forceinline__ float fmax_decode(unsigned k) {
    if (k == 0u) return 0.0f;                       // sentinel: empty segment -> 0 (matches ref)
    unsigned u = (k & 0x80000000u) ? (k & 0x7fffffffu) : ~k;
    return __uint_as_float(u);
}
__device__ __forceinline__ float silu_f(float v) {
    return v * __builtin_amdgcn_rcpf(1.0f + __expf(-v));
}

struct WPtrs { const float* wl[8]; const float* wr[8]; };

// ---------- precompute node-layer B operands in WMMA 16-bit lane layout ----------
__global__ void k_prep_Bnode(WPtrs wp, _Float16* __restrict__ blN) {
    int t = blockIdx.x * blockDim.x + threadIdx.x;      // 8 layers * 32 lanes
    if (t >= 256) return;
    int l = t >> 5, lane = t & 31;
    int D = (l == 0) ? 4 : 8;
    int n = lane & 15, kb = (lane >> 4) * 8;
    const float* Wl = wp.wl[l];
    const float* Wr = wp.wr[l];
#pragma unroll
    for (int j = 0; j < 16; ++j) {
        int K = kb + (j < 8 ? j : j + 8);
        float w = 0.f;
        if (n < 8) {
            if (K < 2 * D)      w = Wl[K * 8 + n];
            else if (K < 3 * D) w = Wr[(K - 2 * D) * 8 + n];
        }
        blN[((size_t)l * 32 + lane) * 16 + j] = (_Float16)w;
    }
}

// ---------- precompute readout B operands (w1: 204x32 -> 7 chunks x 2 halves) ----------
__global__ void k_prep_Bro(const float* __restrict__ w1, _Float16* __restrict__ blR) {
    int t = blockIdx.x * blockDim.x + threadIdx.x;      // 7*2*32 = 448
    if (t >= 448) return;
    int ch = t >> 6, half = (t >> 5) & 1, lane = t & 31;
    int n = lane & 15, kb = (lane >> 4) * 8;
#pragma unroll
    for (int j = 0; j < 16; ++j) {
        int K = ch * 32 + kb + (j < 8 ? j : j + 8);
        float w = (K < ROWW) ? w1[(size_t)K * 32 + half * 16 + n] : 0.f;
        blR[(((size_t)ch * 2 + half) * 32 + lane) * 16 + j] = (_Float16)w;
    }
}

// ---------- init: x (f32,N x4) -> f16 stride-8 features + pool raw input + counts ----------
__global__ void k_init_nodes(const float* __restrict__ x, const int* __restrict__ batch,
                             _Float16* __restrict__ feat, float* __restrict__ gsum,
                             unsigned* __restrict__ gmax, float* __restrict__ cnt, int N_) {
    int n = blockIdx.x * blockDim.x + threadIdx.x;
    if (n >= N_) return;
    float4 v = *(const float4*)(x + (size_t)n * 4);
    v8h h;
    h[0] = (_Float16)v.x; h[1] = (_Float16)v.y; h[2] = (_Float16)v.z; h[3] = (_Float16)v.w;
    h[4] = (_Float16)0.f; h[5] = (_Float16)0.f; h[6] = (_Float16)0.f; h[7] = (_Float16)0.f;
    *(v8h*)(feat + (size_t)n * 8) = h;
    int g = batch[n];
    atomicAdd(cnt + g, 1.0f);
    float* gs = gsum + (size_t)g * POOLW;
    unsigned* gm = gmax + (size_t)g * POOLW;
    atomicAdd(gs + 0, v.x); atomicMax(gm + 0, fmax_key(v.x));
    atomicAdd(gs + 1, v.y); atomicMax(gm + 1, fmax_key(v.y));
    atomicAdd(gs + 2, v.z); atomicMax(gm + 2, fmax_key(v.z));
    atomicAdd(gs + 3, v.w); atomicMax(gm + 3, fmax_key(v.w));
}

// ========== CSR build (once per launch; edge list is layer-invariant) ==========
__global__ void k_count(const int* __restrict__ ei, int* __restrict__ degc, int E_) {
    int e = blockIdx.x * blockDim.x + threadIdx.x;
    if (e >= E_) return;
    atomicAdd(degc + ei[E_ + e], 1);
}

// block-local exclusive scan over 2048 elements (256 thr x 8), emit block sums
__global__ void k_scan1(const int* __restrict__ degc, int* __restrict__ rowptr,
                        int* __restrict__ bsum, int N_) {
    __shared__ int sh[256];
    int t = threadIdx.x;
    int base = blockIdx.x * 2048 + t * 8;
    int loc[8];
    int s = 0;
#pragma unroll
    for (int i = 0; i < 8; ++i) {
        int idx = base + i;
        int xv = (idx < N_) ? degc[idx] : 0;
        loc[i] = s; s += xv;
    }
    sh[t] = s;
    __syncthreads();
    int tot = s;
    for (int off = 1; off < 256; off <<= 1) {
        int xv = (t >= off) ? sh[t - off] : 0;
        __syncthreads();
        sh[t] += xv;
        __syncthreads();
    }
    int texcl = sh[t] - tot;
#pragma unroll
    for (int i = 0; i < 8; ++i) {
        int idx = base + i;
        if (idx < N_) rowptr[idx] = texcl + loc[i];
    }
    if (t == 255) bsum[blockIdx.x] = sh[255];
}

// single-block exclusive scan of the (<=2048) block sums
__global__ void k_scan2(const int* __restrict__ bsum, int* __restrict__ bpre, int nb) {
    __shared__ int sh[256];
    int t = threadIdx.x;
    int base = t * 8;
    int loc[8];
    int s = 0;
#pragma unroll
    for (int i = 0; i < 8; ++i) {
        int xv = (base + i < nb) ? bsum[base + i] : 0;
        loc[i] = s; s += xv;
    }
    sh[t] = s;
    __syncthreads();
    int tot = s;
    for (int off = 1; off < 256; off <<= 1) {
        int xv = (t >= off) ? sh[t - off] : 0;
        __syncthreads();
        sh[t] += xv;
        __syncthreads();
    }
    int texcl = sh[t] - tot;
#pragma unroll
    for (int i = 0; i < 8; ++i) {
        if (base + i < nb) bpre[base + i] = texcl + loc[i];
    }
}

__global__ void k_scan3(int* __restrict__ rowptr, const int* __restrict__ bpre, int N_, int E_) {
    int i = blockIdx.x * blockDim.x + threadIdx.x;
    if (i < N_) rowptr[i] += bpre[i >> 11];
    if (i == 0) rowptr[N_] = E_;
}

__global__ void k_fill(const int* __restrict__ ei, int* __restrict__ cursor,
                       int* __restrict__ col, int E_) {
    int e = blockIdx.x * blockDim.x + threadIdx.x;
    if (e >= E_) return;
    int s = ei[e];
    int t = ei[E_ + e];
    int p = atomicAdd(cursor + t, 1);
    col[p] = s;
}

// ========== fused per-layer kernel: CSR gather (max+mean) -> WMMA -> silu -> pool ==========
// One wave = 16 nodes. Half-wave 0 aggregates MAX (it owns A's K=0..D-1 slots),
// half-wave 1 aggregates SUM->mean (it owns A's K=D..2D-1 slots). No atomics in the
// aggregation; per-graph pooling pre-reduced in LDS (batch is sorted).
template<int D>
__global__ void k_layer_fused(const _Float16* __restrict__ feat, const int* __restrict__ rowptr,
                              const int* __restrict__ col, const int* __restrict__ batch,
                              const _Float16* __restrict__ blN, _Float16* __restrict__ featN,
                              float* __restrict__ gsum, unsigned* __restrict__ gmax,
                              int accOff, int N_, int G_) {
    __shared__ __align__(32) _Float16 sB[512];          // 1 KB: B operand, all 32 lanes
    __shared__ float    psum[GWIN][8];
    __shared__ unsigned pmax[GWIN][8];
    int tid  = threadIdx.x;
    int lane = tid & 31;
    int wib  = tid >> 5;

#if USE_ASYNC_LDS
    {   // stage B layout (256 dwords) global -> LDS via async copy (ASYNCcnt path)
        unsigned ldsOff = (unsigned)(size_t)(&sB[0]) + (unsigned)tid * 4u;
        unsigned vaddr  = (unsigned)tid * 4u;
        asm volatile("global_load_async_to_lds_b32 %0, %1, %2"
                     :: "v"(ldsOff), "v"(vaddr), "s"(blN) : "memory");
        asm volatile("s_wait_asynccnt 0" ::: "memory");
    }
#else
    ((unsigned*)sB)[tid] = ((const unsigned*)blN)[tid];
#endif
    for (int i = tid; i < GWIN * 8; i += 256) {
        ((float*)psum)[i]    = 0.f;
        ((unsigned*)pmax)[i] = 0u;
    }
    __syncthreads();

    int nb0 = blockIdx.x * 128;
    int g0  = batch[nb0 < N_ ? nb0 : (N_ - 1)];

    long nodeBase = (long)nb0 + (long)wib * 16;
    int  m    = lane & 15;
    int  half = lane >> 4;
    int  n    = m;
    long row  = nodeBase + m;
    if (row >= N_) row = N_ - 1;                        // clamp reads; EXEC stays full

    int start = rowptr[row];
    int end   = rowptr[row + 1];
    float rdeg = __builtin_amdgcn_rcpf(fmaxf((float)(end - start), 1.0f));

    v16h a;
    if (D == 8) {
        float a0, a1, a2, a3, a4, a5, a6, a7;
        if (half == 0) {                                // MAX over neighbors (K 0..7)
            a0 = a1 = a2 = a3 = a4 = a5 = a6 = a7 = -3.4e38f;
            for (int p = start; p < end; ++p) {
                v8h f = *(const v8h*)(feat + (size_t)col[p] * 8);
                a0 = fmaxf(a0, (float)f[0]); a1 = fmaxf(a1, (float)f[1]);
                a2 = fmaxf(a2, (float)f[2]); a3 = fmaxf(a3, (float)f[3]);
                a4 = fmaxf(a4, (float)f[4]); a5 = fmaxf(a5, (float)f[5]);
                a6 = fmaxf(a6, (float)f[6]); a7 = fmaxf(a7, (float)f[7]);
            }
            if (end == start) { a0 = a1 = a2 = a3 = a4 = a5 = a6 = a7 = 0.f; }
            a[0] = (_Float16)a0; a[1] = (_Float16)a1; a[2] = (_Float16)a2; a[3] = (_Float16)a3;
            a[4] = (_Float16)a4; a[5] = (_Float16)a5; a[6] = (_Float16)a6; a[7] = (_Float16)a7;
            v8h xr = *(const v8h*)(feat + row * 8);     // K 16..23 = x (raw f16, no cvt)
#pragma unroll
            for (int j = 0; j < 8; ++j) a[8 + j] = xr[j];
        } else {                                        // SUM -> mean (K 8..15)
            a0 = a1 = a2 = a3 = a4 = a5 = a6 = a7 = 0.f;
            for (int p = start; p < end; ++p) {
                v8h f = *(const v8h*)(feat + (size_t)col[p] * 8);
                a0 += (float)f[0]; a1 += (float)f[1]; a2 += (float)f[2]; a3 += (float)f[3];
                a4 += (float)f[4]; a5 += (float)f[5]; a6 += (float)f[6]; a7 += (float)f[7];
            }
            a[0] = (_Float16)(a0 * rdeg); a[1] = (_Float16)(a1 * rdeg);
            a[2] = (_Float16)(a2 * rdeg); a[3] = (_Float16)(a3 * rdeg);
            a[4] = (_Float16)(a4 * rdeg); a[5] = (_Float16)(a5 * rdeg);
            a[6] = (_Float16)(a6 * rdeg); a[7] = (_Float16)(a7 * rdeg);
#pragma unroll
            for (int j = 8; j < 16; ++j) a[j] = (_Float16)0.f;
        }
    } else {                                            // D == 4 (first layer)
        if (half == 0) {                                // K 0..3 = max, K 4..7 = mean
            float m0, m1, m2, m3, s0, s1, s2, s3;
            m0 = m1 = m2 = m3 = -3.4e38f;
            s0 = s1 = s2 = s3 = 0.f;
            for (int p = start; p < end; ++p) {
                v4h f = *(const v4h*)(feat + (size_t)col[p] * 8);
                float f0 = (float)f[0], f1 = (float)f[1], f2 = (float)f[2], f3 = (float)f[3];
                m0 = fmaxf(m0, f0); m1 = fmaxf(m1, f1); m2 = fmaxf(m2, f2); m3 = fmaxf(m3, f3);
                s0 += f0; s1 += f1; s2 += f2; s3 += f3;
            }
            if (end == start) { m0 = m1 = m2 = m3 = 0.f; }
            a[0] = (_Float16)m0; a[1] = (_Float16)m1; a[2] = (_Float16)m2; a[3] = (_Float16)m3;
            a[4] = (_Float16)(s0 * rdeg); a[5] = (_Float16)(s1 * rdeg);
            a[6] = (_Float16)(s2 * rdeg); a[7] = (_Float16)(s3 * rdeg);
#pragma unroll
            for (int j = 8; j < 16; ++j) a[j] = (_Float16)0.f;
        } else {                                        // K 8..11 = x, rest 0
            v4h xr = *(const v4h*)(feat + row * 8);
            a[0] = xr[0]; a[1] = xr[1]; a[2] = xr[2]; a[3] = xr[3];
#pragma unroll
            for (int j = 4; j < 16; ++j) a[j] = (_Float16)0.f;
        }
    }

    v16h b = *(const v16h*)(sB + (size_t)lane * 16);

    v8f c = {};
    c = __builtin_amdgcn_wmma_f32_16x16x32_f16(false, a, false, b, (short)0, c, false, false);

    if (n < 8) {
#pragma unroll
        for (int i = 0; i < 8; ++i) {
            long nd = nodeBase + (i + (half << 3));
            if (nd >= N_) continue;
            float v = silu_f(c[i]);
            featN[nd * 8 + n] = (_Float16)v;
            int g  = batch[nd];
            int gi = g - g0;
            if (gi < GWIN) {
                atomicAdd(&psum[gi][n], v);
                atomicMax(&pmax[gi][n], fmax_key(v));
            } else {
                atomicAdd(gsum + (size_t)g * POOLW + accOff + n, v);
                atomicMax(gmax + (size_t)g * POOLW + accOff + n, fmax_key(v));
            }
        }
    }

    __syncthreads();
    for (int i = tid; i < GWIN * 8; i += 256) {
        int gi = i >> 3, f = i & 7;
        float    s = ((float*)psum)[i];
        unsigned k = ((unsigned*)pmax)[i];
        int g = g0 + gi;
        if (g < G_) {
            if (s != 0.f) atomicAdd(gsum + (size_t)g * POOLW + accOff + f, s);
            if (k != 0u)  atomicMax(gmax + (size_t)g * POOLW + accOff + f, k);
        }
    }
}

// ---------- assemble pooled [G, 224] (padded): per block (mean, max, add) ----------
__global__ void k_pool_finalize(const float* __restrict__ gsum, const unsigned* __restrict__ gmax,
                                const float* __restrict__ cnt, float* __restrict__ pooled, int G_) {
    int idx = blockIdx.x * blockDim.x + threadIdx.x;
    if (idx >= G_ * POOLW) return;
    int g = idx / POOLW, a = idx % POOLW;
    int A0, d;
    if (a < 4) { A0 = 0; d = 4; } else { A0 = 4 + ((a - 4) / 8) * 8; d = 8; }
    int r = a - A0;
    float s  = gsum[idx];
    float c  = fmaxf(cnt[g], 1.0f);
    float mx = fmax_decode(gmax[idx]);
    float* p = pooled + (size_t)g * ROWP + 3 * A0;      // out offset of block == 3*acc offset
    p[r]         = s * __builtin_amdgcn_rcpf(c);        // mean
    p[d + r]     = mx;                                  // max
    p[2 * d + r] = s;                                   // add
}

// ---------- readout GEMM via WMMA: h = silu(pooled[G,224] @ w1[224,32]) ----------
__global__ void k_readout_gemm(const float* __restrict__ pooled, const _Float16* __restrict__ blR,
                               float* __restrict__ h, int G_) {
    int wave = (int)((blockIdx.x * blockDim.x + threadIdx.x) >> 5);
    int lane = threadIdx.x & 31;
    long gBase = (long)wave * 16;
    if (gBase >= G_) return;
    int m  = lane & 15;
    int kb = (lane >> 4) * 8;
    int n  = lane & 15;
    long rowg = gBase + m; if (rowg >= G_) rowg = G_ - 1;

    v8f c0 = {}, c1 = {};
#pragma unroll
    for (int ch = 0; ch < 7; ++ch) {
        const float* pr = pooled + rowg * ROWP + ch * 32 + kb;
        float4 a0 = *(const float4*)(pr);
        float4 a1 = *(const float4*)(pr + 4);
        float4 a2 = *(const float4*)(pr + 16);
        float4 a3 = *(const float4*)(pr + 20);
        v16h a;
        a[0]  = (_Float16)a0.x; a[1]  = (_Float16)a0.y; a[2]  = (_Float16)a0.z; a[3]  = (_Float16)a0.w;
        a[4]  = (_Float16)a1.x; a[5]  = (_Float16)a1.y; a[6]  = (_Float16)a1.z; a[7]  = (_Float16)a1.w;
        a[8]  = (_Float16)a2.x; a[9]  = (_Float16)a2.y; a[10] = (_Float16)a2.z; a[11] = (_Float16)a2.w;
        a[12] = (_Float16)a3.x; a[13] = (_Float16)a3.y; a[14] = (_Float16)a3.z; a[15] = (_Float16)a3.w;
        v16h b0 = *(const v16h*)(blR + (((size_t)ch * 2 + 0) * 32 + lane) * 16);
        v16h b1 = *(const v16h*)(blR + (((size_t)ch * 2 + 1) * 32 + lane) * 16);
        c0 = __builtin_amdgcn_wmma_f32_16x16x32_f16(false, a, false, b0, (short)0, c0, false, false);
        c1 = __builtin_amdgcn_wmma_f32_16x16x32_f16(false, a, false, b1, (short)0, c1, false, false);
    }
#pragma unroll
    for (int i = 0; i < 8; ++i) {
        long g = gBase + (i + ((lane >> 4) << 3));
        if (g >= G_) continue;
        h[g * 32 + n]      = silu_f(c0[i]);
        h[g * 32 + 16 + n] = silu_f(c1[i]);
    }
}

// ---------- per-graph LayerNorm + [32]@[32,2] ----------
__global__ void k_readout_final(const float* __restrict__ h, const float* __restrict__ lnw,
                                const float* __restrict__ lnb, const float* __restrict__ w2,
                                float* __restrict__ out, int G_) {
    int g = blockIdx.x * blockDim.x + threadIdx.x;
    if (g >= G_) return;
    float v[32];
    float mu = 0.f;
#pragma unroll
    for (int i = 0; i < 32; ++i) { v[i] = h[(size_t)g * 32 + i]; mu += v[i]; }
    mu *= (1.0f / 32.0f);
    float var = 0.f;
#pragma unroll
    for (int i = 0; i < 32; ++i) { float t = v[i] - mu; var += t * t; }
    var *= (1.0f / 32.0f);
    float inv = rsqrtf(var + 1e-5f);
    float o0 = 0.f, o1 = 0.f;
#pragma unroll
    for (int i = 0; i < 32; ++i) {
        float t = (v[i] - mu) * inv * lnw[i] + lnb[i];
        o0 += t * w2[i * 2 + 0];
        o1 += t * w2[i * 2 + 1];
    }
    out[(size_t)g * 2 + 0] = o0;
    out[(size_t)g * 2 + 1] = o1;
}

extern "C" void kernel_launch(void* const* d_in, const int* in_sizes, int n_in,
                              void* d_out, int out_size, void* d_ws, size_t ws_size,
                              hipStream_t stream) {
    const float* x     = (const float*)d_in[0];
    const int*   ei    = (const int*)d_in[1];
    const int*   batch = (const int*)d_in[2];
    WPtrs wp;
    for (int l = 0; l < 8; ++l) { wp.wl[l] = (const float*)d_in[3 + l]; wp.wr[l] = (const float*)d_in[11 + l]; }
    const float* w1  = (const float*)d_in[19];
    const float* lnw = (const float*)d_in[20];
    const float* lnb = (const float*)d_in[21];
    const float* w2  = (const float*)d_in[22];
    float* out = (float*)d_out;

    const int N_ = in_sizes[2];          // batch has N entries
    const int E_ = in_sizes[1] / 2;      // edge_index is (2, E)
    const int G_ = out_size / 2;         // output is (G, 2)
    const int nb = (N_ + 2047) / 2048;   // scan blocks (<= 2048 supported)

    // workspace carve-up (256B aligned slabs)
    char* ws = (char*)d_ws;
    auto take = [&](size_t bytes) { void* p = ws; ws += (bytes + 255) & ~(size_t)255; return p; };
    _Float16*  feat0  = (_Float16*)take((size_t)N_ * 8 * sizeof(_Float16));
    _Float16*  feat1  = (_Float16*)take((size_t)N_ * 8 * sizeof(_Float16));
    int*       degc   = (int*)take((size_t)N_ * sizeof(int));
    int*       rowptr = (int*)take(((size_t)N_ + 1) * sizeof(int));
    int*       cursor = (int*)take((size_t)N_ * sizeof(int));
    int*       colbuf = (int*)take((size_t)E_ * sizeof(int));
    int*       bsum   = (int*)take((size_t)nb * sizeof(int));
    int*       bpre   = (int*)take((size_t)nb * sizeof(int));
    float*     cnt    = (float*)take((size_t)G_ * sizeof(float));
    float*     gsum   = (float*)take((size_t)G_ * POOLW * sizeof(float));
    unsigned*  gmax   = (unsigned*)take((size_t)G_ * POOLW * sizeof(unsigned));
    float*     pooled = (float*)take((size_t)G_ * ROWP * sizeof(float));
    float*     hbuf   = (float*)take((size_t)G_ * 32 * sizeof(float));
    _Float16*  blN    = (_Float16*)take(8 * 32 * 16 * sizeof(_Float16));
    _Float16*  blR    = (_Float16*)take(14 * 32 * 16 * sizeof(_Float16));

    const int B = 256;
    // init + weight pre-layout
    hipMemsetAsync(degc,   0, (size_t)N_ * sizeof(int), stream);
    hipMemsetAsync(cnt,    0, (size_t)G_ * sizeof(float), stream);
    hipMemsetAsync(gsum,   0, (size_t)G_ * POOLW * sizeof(float), stream);
    hipMemsetAsync(gmax,   0, (size_t)G_ * POOLW * sizeof(unsigned), stream);
    hipMemsetAsync(pooled, 0, (size_t)G_ * ROWP * sizeof(float), stream);
    k_prep_Bnode<<<1, 256, 0, stream>>>(wp, blN);
    k_prep_Bro<<<2, 256, 0, stream>>>(w1, blR);
    k_init_nodes<<<(N_ + B - 1) / B, B, 0, stream>>>(x, batch, feat0, gsum, gmax, cnt, N_);

    // CSR build (counting sort by dst), once per launch
    k_count<<<(E_ + B - 1) / B, B, 0, stream>>>(ei, degc, E_);
    k_scan1<<<nb, 256, 0, stream>>>(degc, rowptr, bsum, N_);
    k_scan2<<<1, 256, 0, stream>>>(bsum, bpre, nb);
    k_scan3<<<(N_ + B - 1) / B, B, 0, stream>>>(rowptr, bpre, N_, E_);
    hipMemcpyAsync(cursor, rowptr, (size_t)N_ * sizeof(int), hipMemcpyDeviceToDevice, stream);
    k_fill<<<(E_ + B - 1) / B, B, 0, stream>>>(ei, cursor, colbuf, E_);

    // 8 fused SAGE layers: gather-aggregate -> WMMA -> silu -> pool (no atomics in hot loop)
    _Float16* cur = feat0;
    _Float16* nxt = feat1;
    const int nodeBlocks = (N_ + 127) / 128;   // 8 waves/block, 16 nodes/wave
    for (int l = 0; l < 8; ++l) {
        int accOff = 4 + l * 8;
        if (l == 0) {
            k_layer_fused<4><<<nodeBlocks, B, 0, stream>>>(cur, rowptr, colbuf, batch,
                                                           blN + (size_t)l * 512, nxt,
                                                           gsum, gmax, accOff, N_, G_);
        } else {
            k_layer_fused<8><<<nodeBlocks, B, 0, stream>>>(cur, rowptr, colbuf, batch,
                                                           blN + (size_t)l * 512, nxt,
                                                           gsum, gmax, accOff, N_, G_);
        }
        _Float16* t = cur; cur = nxt; nxt = t;
    }

    k_pool_finalize<<<(G_ * POOLW + B - 1) / B, B, 0, stream>>>(gsum, gmax, cnt, pooled, G_);
    const int gtiles = (G_ + 15) / 16;
    k_readout_gemm<<<(gtiles * 32 + B - 1) / B, B, 0, stream>>>(pooled, blR, hbuf, G_);
    k_readout_final<<<(G_ + B - 1) / B, B, 0, stream>>>(hbuf, lnw, lnb, w2, out, G_);
}
